// LSTMRNN_62577673503878
// MI455X (gfx1250) — matrix operand order repaired and checked by reading, not compiled, not run on hardware
//
#include <hip/hip_runtime.h>
#include <hip/hip_bf16.h>

#define H 1024
#define G4 4096      // 4*H
#define TSTEPS 64
#define VOCABN 32000

typedef __attribute__((ext_vector_type(16))) __bf16 v16bf;
typedef __attribute__((ext_vector_type(8)))  float  v8f;

template<bool RELU>
__device__ inline void put4(v16bf& v, int o, float4 q)
{
    if (RELU) {
        q.x = fmaxf(q.x, 0.f); q.y = fmaxf(q.y, 0.f);
        q.z = fmaxf(q.z, 0.f); q.w = fmaxf(q.w, 0.f);
    }
    v[o]     = (__bf16)q.x;
    v[o + 1] = (__bf16)q.y;
    v[o + 2] = (__bf16)q.z;
    v[o + 3] = (__bf16)q.w;
}

// ---------------------------------------------------------------------------
// WMMA GEMM:  C[M,N] = op(A)[M,K] @ W[N,K]^T  (+ bias0[n] + bias1[n])
// M fixed at 64 (4 tiles). One wave owns one 16-wide N strip and computes all
// 4 M-tiles, reusing the B fragment across 4 consecutive WMMAs so each row of
// W is fetched from HBM exactly once.
// ---------------------------------------------------------------------------
template<bool RELU>
__global__ __launch_bounds__(256) void gemm_wmma_kernel(
    const float* __restrict__ A, const int* __restrict__ idx,
    int shift_bos,
    const float* __restrict__ W,
    const float* __restrict__ bias0, const float* __restrict__ bias1,
    float* __restrict__ C, int N, int K)
{
    const int lane = threadIdx.x & 31;
    const int half = lane >> 4;       // 0: lanes 0-15, 1: lanes 16-31
    const int l15  = lane & 15;

    const int ntn = N >> 4;
    const int tn  = blockIdx.x * (blockDim.x >> 5) + (threadIdx.x >> 5);
    if (tn >= ntn) return;

    const float* ArowP[4];
#pragma unroll
    for (int tm = 0; tm < 4; ++tm) {
        const int arow = tm * 16 + l15;
        if (idx) {
            int tok = shift_bos ? (arow == 0 ? 0 : idx[arow - 1]) : idx[arow];
            ArowP[tm] = A + (size_t)tok * K;
        } else {
            ArowP[tm] = A + (size_t)arow * K;
        }
    }
    const float* Brow = W + (size_t)(tn * 16 + l15) * K;

    v8f acc[4];
#pragma unroll
    for (int tm = 0; tm < 4; ++tm)
        acc[tm] = (v8f){0.f, 0.f, 0.f, 0.f, 0.f, 0.f, 0.f, 0.f};

    for (int kk = 0; kk < K; kk += 32) {
        const float4* pb4 = (const float4*)(Brow + kk + half * 8);
        float4 braw0 = pb4[0], braw1 = pb4[1], braw2 = pb4[4], braw3 = pb4[5];

        float4 araw[4][4];
#pragma unroll
        for (int tm = 0; tm < 4; ++tm) {
            const float4* pa0 = (const float4*)(ArowP[tm] + kk + half * 8);
            const float4* pa1 = (const float4*)(ArowP[tm] + kk + 16 + half * 8);
            araw[tm][0] = pa0[0];
            araw[tm][1] = pa0[1];
            araw[tm][2] = pa1[0];
            araw[tm][3] = pa1[1];
        }

        v16bf b;
        put4<false>(b, 0, braw0);  put4<false>(b, 4, braw1);
        put4<false>(b, 8, braw2);  put4<false>(b, 12, braw3);

        v16bf afr[4];
#pragma unroll
        for (int tm = 0; tm < 4; ++tm) {
            put4<RELU>(afr[tm], 0,  araw[tm][0]);
            put4<RELU>(afr[tm], 4,  araw[tm][1]);
            put4<RELU>(afr[tm], 8,  araw[tm][2]);
            put4<RELU>(afr[tm], 12, araw[tm][3]);
        }

#pragma unroll
        for (int tm = 0; tm < 4; ++tm)
            acc[tm] = __builtin_amdgcn_wmma_f32_16x16x32_bf16(
                false, afr[tm], false, b, (short)0, acc[tm], false, false);
    }

    const int n = tn * 16 + l15;
    const float badd = (bias0 ? bias0[n] : 0.f) + (bias1 ? bias1[n] : 0.f);
#pragma unroll
    for (int tm = 0; tm < 4; ++tm) {
#pragma unroll
        for (int r = 0; r < 8; ++r) {
            const int m = tm * 16 + r + half * 8;
            C[(size_t)m * N + n] = acc[tm][r] + badd;
        }
    }
}

// ---------------------------------------------------------------------------
// One-time Whh f32 -> packed bf16 (RNE), halves the scan's per-step traffic
// ---------------------------------------------------------------------------
__device__ inline unsigned bf16rne(float x)
{
    unsigned u = __float_as_uint(x);
    return (u + 0x7fffu + ((u >> 16) & 1u)) >> 16;
}
__device__ inline unsigned packbf(float a, float b)
{
    return bf16rne(a) | (bf16rne(b) << 16);
}

__global__ __launch_bounds__(256) void wconv_kernel(
    const float* __restrict__ src, uint2* __restrict__ dst, int n4)
{
    int i = blockIdx.x * blockDim.x + threadIdx.x;
    if (i >= n4) return;
    float4 q = ((const float4*)src)[i];
    dst[i] = make_uint2(packbf(q.x, q.y), packbf(q.z, q.w));
}

// ---------------------------------------------------------------------------
// Recurrent LSTM scan (persistent cooperative, one grid barrier per step)
// ---------------------------------------------------------------------------
__device__ inline float sigf(float x)    { return 1.f / (1.f + __expf(-x)); }
__device__ inline float tanhfast(float x){ return 2.f / (1.f + __expf(-2.f * x)) - 1.f; }
__device__ inline float bflo(unsigned u) { return __uint_as_float(u << 16); }
__device__ inline float bfhi(unsigned u) { return __uint_as_float(u & 0xffff0000u); }

__device__ inline void grid_barrier(unsigned* bar, unsigned target, unsigned nwg)
{
    __threadfence();
    __syncthreads();
    if (threadIdx.x == 0) {
        unsigned prev = __hip_atomic_fetch_add(bar, 1u, __ATOMIC_ACQ_REL,
                                               __HIP_MEMORY_SCOPE_AGENT);
        if (prev + 1 == target * nwg) {
            __hip_atomic_store(bar + 1, target, __ATOMIC_RELEASE,
                               __HIP_MEMORY_SCOPE_AGENT);
        } else {
            while (__hip_atomic_load(bar + 1, __ATOMIC_ACQUIRE,
                                     __HIP_MEMORY_SCOPE_AGENT) < target)
                __builtin_amdgcn_s_sleep(1);
        }
    }
    __syncthreads();
}

template<bool WBF16>
__device__ inline float dot_row(const void* Whh, int row, const float* hs, int lane)
{
    float acc = 0.f;
    if (WBF16) {
        const uint4*  Wr = (const uint4*)((const unsigned short*)Whh + (size_t)row * H);
        const float4* Hv = (const float4*)hs;
#pragma unroll
        for (int it = 0; it < 4; ++it) {
            uint4  wq = Wr[lane + 32 * it];
            float4 ha = Hv[2 * (lane + 32 * it)];
            float4 hb = Hv[2 * (lane + 32 * it) + 1];
            acc = fmaf(bflo(wq.x), ha.x, acc); acc = fmaf(bfhi(wq.x), ha.y, acc);
            acc = fmaf(bflo(wq.y), ha.z, acc); acc = fmaf(bfhi(wq.y), ha.w, acc);
            acc = fmaf(bflo(wq.z), hb.x, acc); acc = fmaf(bfhi(wq.z), hb.y, acc);
            acc = fmaf(bflo(wq.w), hb.z, acc); acc = fmaf(bfhi(wq.w), hb.w, acc);
        }
    } else {
        const float4* Wr = (const float4*)((const float*)Whh + (size_t)row * H);
        const float4* Hv = (const float4*)hs;
#pragma unroll
        for (int it = 0; it < 8; ++it) {
            float4 w  = Wr[lane + 32 * it];
            float4 hv = Hv[lane + 32 * it];
            acc = fmaf(w.x, hv.x, acc);
            acc = fmaf(w.y, hv.y, acc);
            acc = fmaf(w.z, hv.z, acc);
            acc = fmaf(w.w, hv.w, acc);
        }
    }
    return acc;
}

// 64 WGs x 256 threads per scan slot; wave w owns gate indices j=2w, 2w+1.
// c-state lives in registers; h double-buffered in global and staged per-step
// into LDS. Whh streams from the 192 MB L2.
template<bool WBF16>
__device__ inline void scan_body(
    int localwg, unsigned NWG, float* hs,
    const float* __restrict__ X, const void* __restrict__ Whh,
    const float* __restrict__ init_h, const float* __restrict__ init_c,
    float* __restrict__ hA, float* __restrict__ hB,
    float* __restrict__ Hseq,
    float* __restrict__ carry_h, float* __restrict__ carry_c,
    unsigned* bar)
{
    const int lane  = threadIdx.x & 31;
    const int gwave = localwg * 8 + (int)(threadIdx.x >> 5);
    const int j0    = gwave * 2;

    float c0 = init_c[j0], c1 = init_c[j0 + 1];
    float h0 = init_h[j0], h1 = init_h[j0 + 1];
    if (lane == 0) hA[j0]     = h0;
    if (lane == 1) hA[j0 + 1] = h1;
    grid_barrier(bar, 1, NWG);

    float* hcur  = hA;
    float* hnext = hB;

    for (int t = 0; t < TSTEPS; ++t) {
        ((float4*)hs)[threadIdx.x] = ((const float4*)hcur)[threadIdx.x];
        __syncthreads();

        const float* Xt = X + (size_t)t * G4;
        float g[8];
#pragma unroll
        for (int jj = 0; jj < 2; ++jj) {
#pragma unroll
            for (int gt = 0; gt < 4; ++gt) {
                float acc = dot_row<WBF16>(Whh, gt * H + j0 + jj, hs, lane);
#pragma unroll
                for (int off = 16; off > 0; off >>= 1)
                    acc += __shfl_xor(acc, off, 32);   // all lanes get total
                g[jj * 4 + gt] = acc;
            }
        }
        {   // gate order i, f, g, o (rows j, j+H, j+2H, j+3H)
            const int j = j0;
            float gi = g[0] + Xt[j];
            float gf = g[1] + Xt[H + j];
            float gz = g[2] + Xt[2 * H + j];
            float go = g[3] + Xt[3 * H + j];
            c0 = sigf(gf) * c0 + sigf(gi) * tanhfast(gz);
            h0 = sigf(go) * tanhfast(c0);
        }
        {
            const int j = j0 + 1;
            float gi = g[4] + Xt[j];
            float gf = g[5] + Xt[H + j];
            float gz = g[6] + Xt[2 * H + j];
            float go = g[7] + Xt[3 * H + j];
            c1 = sigf(gf) * c1 + sigf(gi) * tanhfast(gz);
            h1 = sigf(go) * tanhfast(c1);
        }
        if (lane == 0) hnext[j0]     = h0;
        if (lane == 1) hnext[j0 + 1] = h1;
        if (Hseq) {
            if (lane == 0) Hseq[(size_t)t * H + j0]     = h0;
            if (lane == 1) Hseq[(size_t)t * H + j0 + 1] = h1;
        }
        grid_barrier(bar, (unsigned)(t + 2), NWG);
        float* tmp = hcur; hcur = hnext; hnext = tmp;
    }

    if (lane == 0) { carry_h[j0]     = h0; carry_c[j0]     = c0; }
    if (lane == 1) { carry_h[j0 + 1] = h1; carry_c[j0 + 1] = c1; }
}

template<bool WBF16>
__global__ __launch_bounds__(256) void lstm_scan_kernel(
    const float* X, const void* Whh,
    const float* init_h, const float* init_c,
    float* hA, float* hB, float* Hseq,
    float* carry_h, float* carry_c, unsigned* bar)
{
    __shared__ float hs[H];
    scan_body<WBF16>(blockIdx.x, 64u, hs, X, Whh, init_h, init_c,
                     hA, hB, Hseq, carry_h, carry_c, bar);
}

// Two independent scans fused in one 128-WG launch (enc layer1 || dec layer0):
// dec0 depends only on embeddings + enc0's final carry, not on enc1.
template<bool WBF16>
__global__ __launch_bounds__(256) void lstm_scan_dual_kernel(
    const float* Xa, const void* Wa, const float* iha, const float* ica,
    float* hAa, float* hBa, float* Hseqa, float* cha, float* cca, unsigned* bara,
    const float* Xb, const void* Wb, const float* ihb, const float* icb,
    float* hAb, float* hBb, float* Hseqb, float* chb, float* ccb, unsigned* barb)
{
    __shared__ float hs[H];
    if (blockIdx.x < 64)
        scan_body<WBF16>(blockIdx.x, 64u, hs, Xa, Wa, iha, ica,
                         hAa, hBa, Hseqa, cha, cca, bara);
    else
        scan_body<WBF16>(blockIdx.x - 64, 64u, hs, Xb, Wb, ihb, icb,
                         hAb, hBb, Hseqb, chb, ccb, barb);
}

// ---------------------------------------------------------------------------
extern "C" void kernel_launch(void* const* d_in, const int* in_sizes, int n_in,
                              void* d_out, int out_size, void* d_ws, size_t ws_size,
                              hipStream_t stream)
{
    const int*   src_tok  = (const int*)  d_in[0];
    const int*   tgt_tok  = (const int*)  d_in[1];
    const float* emb_src  = (const float*)d_in[2];
    const float* emb_tgt  = (const float*)d_in[3];
    const float* eWih0 = (const float*)d_in[4],  *eWhh0 = (const float*)d_in[5];
    const float* ebih0 = (const float*)d_in[6],  *ebhh0 = (const float*)d_in[7];
    const float* eWih1 = (const float*)d_in[8],  *eWhh1 = (const float*)d_in[9];
    const float* ebih1 = (const float*)d_in[10], *ebhh1 = (const float*)d_in[11];
    const float* dWih0 = (const float*)d_in[12], *dWhh0 = (const float*)d_in[13];
    const float* dbih0 = (const float*)d_in[14], *dbhh0 = (const float*)d_in[15];
    const float* dWih1 = (const float*)d_in[16], *dWhh1 = (const float*)d_in[17];
    const float* dbih1 = (const float*)d_in[18], *dbhh1 = (const float*)d_in[19];
    const float* out_W = (const float*)d_in[20];
    const float* out_b = (const float*)d_in[21];
    float* out = (float*)d_out;

    // ---- workspace layout (floats) ----------------------------------------
    float* wsf  = (float*)d_ws;
    float* zero = wsf;                     // 2048 (zero init h/c)
    float* Xe   = zero + 2048;             // 64*4096 (enc/dec1 projections)
    float* Xd   = Xe   + 64 * G4;          // 64*4096 (dec0 projection)
    float* Hs0  = Xd   + 64 * G4;          // 64*1024 enc0 hidden seq
    float* Hs1  = Hs0  + 64 * H;           // 64*1024 dec0 hidden seq
    float* Hs2  = Hs1  + 64 * H;           // 64*1024 dec1 hidden seq
    float* hPA  = Hs2  + 64 * H;           // 4 x 1024 h ping-pong
    float* hPB  = hPA  + H;
    float* hPC  = hPB  + H;
    float* hPD  = hPC  + H;
    float* carry = hPD + H;                // 4 x {h:1024, c:1024}
    unsigned* bars = (unsigned*)(carry + 8192);   // 4 x {count, epoch} + pad

    // optional packed-bf16 Whh copies (4 x 8 MB), 256B aligned
    char* wb = (char*)(bars + 16);
    wb += (256 - ((size_t)wb & 255)) & 255;
    unsigned short* Wb[4];
    const size_t welems = (size_t)G4 * H;
    for (int i = 0; i < 4; ++i) Wb[i] = (unsigned short*)(wb + (size_t)i * welems * 2);
    const size_t need_bf16 = (size_t)(wb - (char*)d_ws) + 4 * welems * 2;
    const bool usebf16 = (ws_size >= need_bf16);

    hipMemsetAsync(zero, 0, 2048 * sizeof(float), stream);
    hipMemsetAsync(bars, 0, 16 * sizeof(unsigned), stream);

    const int gemmBlocksSmall = ((G4 / 16) + 7) / 8;        // 32
    const int gemmBlocksOut   = ((VOCABN / 16) + 7) / 8;    // 250
    const int wconvBlocks     = (int)(welems / 4 / 256);    // 4096

    const void* W0 = usebf16 ? (const void*)Wb[0] : (const void*)eWhh0;
    const void* W1 = usebf16 ? (const void*)Wb[1] : (const void*)eWhh1;
    const void* W2 = usebf16 ? (const void*)Wb[2] : (const void*)dWhh0;
    const void* W3 = usebf16 ? (const void*)Wb[3] : (const void*)dWhh1;

    if (usebf16) {
        wconv_kernel<<<wconvBlocks, 256, 0, stream>>>(eWhh0, (uint2*)Wb[0], (int)(welems / 4));
        wconv_kernel<<<wconvBlocks, 256, 0, stream>>>(eWhh1, (uint2*)Wb[1], (int)(welems / 4));
        wconv_kernel<<<wconvBlocks, 256, 0, stream>>>(dWhh0, (uint2*)Wb[2], (int)(welems / 4));
        wconv_kernel<<<wconvBlocks, 256, 0, stream>>>(dWhh1, (uint2*)Wb[3], (int)(welems / 4));
    }

    // input projections that depend only on embeddings
    gemm_wmma_kernel<false><<<gemmBlocksSmall, 256, 0, stream>>>(
        emb_src, src_tok, 0, eWih0, ebih0, ebhh0, Xe, G4, H);
    gemm_wmma_kernel<false><<<gemmBlocksSmall, 256, 0, stream>>>(
        emb_tgt, tgt_tok, 1, dWih0, dbih0, dbhh0, Xd, G4, H);

    // encoder layer 0 scan
    if (usebf16)
        lstm_scan_kernel<true><<<64, 256, 0, stream>>>(
            Xe, W0, zero, zero + H, hPA, hPB, Hs0, carry + 0, carry + H, bars + 0);
    else
        lstm_scan_kernel<false><<<64, 256, 0, stream>>>(
            Xe, W0, zero, zero + H, hPA, hPB, Hs0, carry + 0, carry + H, bars + 0);

    // encoder layer 1 projection
    gemm_wmma_kernel<false><<<gemmBlocksSmall, 256, 0, stream>>>(
        Hs0, nullptr, 0, eWih1, ebih1, ebhh1, Xe, G4, H);

    // fused: encoder layer 1 scan || decoder layer 0 scan
    if (usebf16)
        lstm_scan_dual_kernel<true><<<128, 256, 0, stream>>>(
            Xe, W1, zero, zero + H, hPA, hPB, nullptr,
            carry + 2048, carry + 2048 + H, bars + 2,
            Xd, W2, carry + 0, carry + H, hPC, hPD, Hs1,
            carry + 4096, carry + 4096 + H, bars + 4);
    else
        lstm_scan_dual_kernel<false><<<128, 256, 0, stream>>>(
            Xe, W1, zero, zero + H, hPA, hPB, nullptr,
            carry + 2048, carry + 2048 + H, bars + 2,
            Xd, W2, carry + 0, carry + H, hPC, hPD, Hs1,
            carry + 4096, carry + 4096 + H, bars + 4);

    // decoder layer 1
    gemm_wmma_kernel<false><<<gemmBlocksSmall, 256, 0, stream>>>(
        Hs1, nullptr, 0, dWih1, dbih1, dbhh1, Xe, G4, H);
    if (usebf16)
        lstm_scan_kernel<true><<<64, 256, 0, stream>>>(
            Xe, W3, carry + 2048, carry + 2048 + H, hPA, hPB, Hs2,
            carry + 6144, carry + 6144 + H, bars + 6);
    else
        lstm_scan_kernel<false><<<64, 256, 0, stream>>>(
            Xe, W3, carry + 2048, carry + 2048 + H, hPA, hPB, Hs2,
            carry + 6144, carry + 6144 + H, bars + 6);

    // logits = relu(Hdec1) @ out_W^T + out_b   (64 x 32000)
    gemm_wmma_kernel<true><<<gemmBlocksOut, 256, 0, stream>>>(
        Hs2, nullptr, 0, out_W, out_b, nullptr, out, VOCABN, H);
}
